// BrainMT_74088185856519
// MI455X (gfx1250) — matrix operand order repaired and checked by reading, not compiled, not run on hardware
//
#include <hip/hip_runtime.h>
#include <hip/hip_bf16.h>
#include <stdint.h>

typedef __attribute__((ext_vector_type(16))) _Float16 v16h;
typedef __attribute__((ext_vector_type(8)))  _Float16 h8;
typedef __attribute__((ext_vector_type(8)))  float    v8f;
typedef __attribute__((ext_vector_type(4)))  float    fx4;

// ---------------- device helpers ----------------
__device__ __forceinline__ float dev_gelu(float v) {
  float t = tanhf(0.7978845608028654f * (v + 0.044715f * v * v * v));
  return 0.5f * v * (1.0f + t);
}
__device__ __forceinline__ float dev_softplus(float v) {
  return (v > 20.f) ? v : log1pf(expf(v));
}
__device__ __forceinline__ float dev_silu(float v) { return v / (1.f + expf(-v)); }

// ---------------- generic WMMA GEMM ----------------
enum { GF_AHALF=1, GF_BHALF=2, GF_BKMAJ=4, GF_BIAS=8, GF_RELU=16, GF_GELU=32,
       GF_ADDTO=64, GF_SOFTPLUS=128 };

struct GemmArgs {
  const void* A; const void* B; float* C; const float* bias;
  int M, N, Kb;            // Kb = K per tap (Cin for conv, K for linear)
  int lda, ldb, ldc;
  int batch, binner;
  long sAo, sAi, sBo, sBi, sCo, sCi;
  float scale;
  int flags;
  int conv;                // 0 = linear, 1 = 3x3x3 conv (27 taps)
  int frames, ID, IH, IW, OD, OH, OW, cstride;
};

template<int AHALF, int BHALF, int BKMAJ>
__global__ __launch_bounds__(256) void gemm_wmma_kernel(GemmArgs g) {
  int wid  = threadIdx.x >> 5;
  int lane = threadIdx.x & 31;
  long tilesM = (g.M + 15) >> 4;
  long tilesN = (g.N + 15) >> 4;
  long tpb  = tilesM * tilesN;
  long tile = (long)blockIdx.x * 8 + wid;
  if (tile >= tpb * g.batch) return;              // wave-uniform exit
  int  z  = (int)(tile / tpb);
  long rr = tile % tpb;
  int  mt = (int)(rr / tilesN);
  int  nt = (int)(rr % tilesN);
  int  bo = z / g.binner, bi = z % g.binner;
  long offA = (long)bo * g.sAo + (long)bi * g.sAi;
  long offB = (long)bo * g.sBo + (long)bi * g.sBi;
  long offC = (long)bo * g.sCo + (long)bi * g.sCi;

  int m0 = mt << 4, n0 = nt << 4;
  int ml  = lane & 15;
  int kbA = (lane >> 4) << 4;    // contiguous k base: 0 / 16
  int kbB = (lane >> 4) << 4;    // contiguous k base: 0 / 16
  int nB  = n0 + ml;

  v8f acc = {0.f,0.f,0.f,0.f,0.f,0.f,0.f,0.f};
  int taps = g.conv ? 27 : 1;
  for (int t = 0; t < taps; ++t) {
    bool rv = false; long rowBase = 0;
    int mg = m0 + ml;
    if (g.conv) {
      if (mg < g.M) {
        int OSP = g.OD * g.OH * g.OW;
        int f  = mg / OSP;
        int rp = mg - f * OSP;
        int od = rp / (g.OH * g.OW);
        int r2 = rp - od * g.OH * g.OW;
        int oh = r2 / g.OW;
        int ow = r2 - oh * g.OW;
        int kd = t / 9, kh = (t / 3) % 3, kw = t % 3;
        int id = od * g.cstride + kd - 1;
        int ih = oh * g.cstride + kh - 1;
        int iw = ow * g.cstride + kw - 1;
        if ((unsigned)id < (unsigned)g.ID && (unsigned)ih < (unsigned)g.IH &&
            (unsigned)iw < (unsigned)g.IW) {
          rv = true;
          rowBase = ((long)((f * g.ID + id) * g.IH + ih) * g.IW + iw) * (long)g.lda;
        }
      }
    } else {
      rv = (mg < g.M);
      rowBase = (long)mg * g.lda;
    }
    long bTap = g.conv ? (long)t * (long)g.N * (long)g.Kb : 0;
    for (int k0 = 0; k0 < g.Kb; k0 += 32) {
      bool fullK = (k0 + 32 <= g.Kb);
      v16h a, b;
      // ---------- A fragment (lanes 0-15 K=0..15, 16-31 K=16..31) ----------
      if (rv && fullK) {
        if (AHALF) {
          const h8* p = (const h8*)((const _Float16*)g.A + offA + rowBase + k0 + kbA);
          h8 lo = p[0], hi = p[1];            // K = kbA+0..15
          #pragma unroll
          for (int e = 0; e < 8; ++e) { a[e] = lo[e]; a[e + 8] = hi[e]; }
          __builtin_prefetch((const _Float16*)g.A + offA + rowBase + k0 + 64, 0, 1);
        } else {
          const fx4* p = (const fx4*)((const float*)g.A + offA + rowBase + k0 + kbA);
          fx4 q0 = p[0], q1 = p[1], q2 = p[2], q3 = p[3];
          #pragma unroll
          for (int e = 0; e < 4; ++e) {
            a[e]      = (_Float16)q0[e];
            a[e + 4]  = (_Float16)q1[e];
            a[e + 8]  = (_Float16)q2[e];
            a[e + 12] = (_Float16)q3[e];
          }
          __builtin_prefetch((const float*)g.A + offA + rowBase + k0 + 64, 0, 1);
        }
      } else {
        #pragma unroll
        for (int e = 0; e < 16; ++e) {
          int k = k0 + kbA + e;
          _Float16 hv = (_Float16)0.f;
          if (rv && k < g.Kb) {
            if (AHALF) hv = ((const _Float16*)g.A)[offA + rowBase + k];
            else       hv = (_Float16)((const float*)g.A)[offA + rowBase + k];
          }
          a[e] = hv;
        }
      }
      // ---------- B fragment (lanes 0-15 K=0..15, 16-31 K=16..31) ----------
      if (BKMAJ) {
        #pragma unroll
        for (int e = 0; e < 16; ++e) {
          int k = k0 + kbB + e;
          _Float16 hv = (_Float16)0.f;
          if (k < g.Kb && nB < g.N)
            hv = (_Float16)((const float*)g.B)[offB + (long)k * g.ldb + nB];
          b[e] = hv;
        }
      } else if (fullK && nB < g.N) {
        if (BHALF) {
          const h8* p = (const h8*)((const _Float16*)g.B + offB + bTap +
                                    (long)nB * g.ldb + k0 + kbB);
          h8 b0 = p[0], b1 = p[1];
          #pragma unroll
          for (int e = 0; e < 8; ++e) { b[e] = b0[e]; b[e + 8] = b1[e]; }
        } else {
          const fx4* p = (const fx4*)((const float*)g.B + offB + bTap +
                                      (long)nB * g.ldb + k0 + kbB);
          fx4 q0 = p[0], q1 = p[1], q2 = p[2], q3 = p[3];
          #pragma unroll
          for (int e = 0; e < 4; ++e) {
            b[e]      = (_Float16)q0[e];
            b[e + 4]  = (_Float16)q1[e];
            b[e + 8]  = (_Float16)q2[e];
            b[e + 12] = (_Float16)q3[e];
          }
        }
      } else {
        #pragma unroll
        for (int e = 0; e < 16; ++e) {
          int k = k0 + kbB + e;
          _Float16 hv = (_Float16)0.f;
          if (k < g.Kb && nB < g.N) {
            long bix = offB + bTap + (long)nB * g.ldb + k;
            if (BHALF) hv = ((const _Float16*)g.B)[bix];
            else       hv = (_Float16)((const float*)g.B)[bix];
          }
          b[e] = hv;
        }
      }
      acc = __builtin_amdgcn_wmma_f32_16x16x32_f16(false, a, false, b,
                                                   (short)0, acc, false, false);
    }
  }
  // epilogue (C layout: VGPR r <-> M = r + 8*(lane>=16), N = lane&15)
  int ncol = n0 + ml;
  if (ncol >= g.N) return;
  float bv = (g.flags & GF_BIAS) ? g.bias[ncol] : 0.f;
  int mbase = m0 + ((lane >> 4) << 3);
  #pragma unroll
  for (int r = 0; r < 8; ++r) {
    int m = mbase + r;
    if (m < g.M) {
      float v = acc[r] * g.scale + bv;
      if (g.flags & GF_GELU)     v = dev_gelu(v);
      if (g.flags & GF_RELU)     v = fmaxf(v, 0.f);
      if (g.flags & GF_SOFTPLUS) v = dev_softplus(v);
      long ci = offC + (long)m * g.ldc + ncol;
      if (g.flags & GF_ADDTO) v += g.C[ci];
      g.C[ci] = v;
    }
  }
}

// ---------------- LayerNorm / RMSNorm (wave per row, wave32) ----------------
enum { LF_RMS=1, LF_GELU=2, LF_RES=4, LF_BIAS=8 };
struct LnArgs { const float* X; const float* W; const float* Bp; const float* Res;
                float* O; int rows; int C; int flags; };

__global__ __launch_bounds__(256) void ln_kernel(LnArgs a) {
  int wid = threadIdx.x >> 5, lane = threadIdx.x & 31;
  int row = blockIdx.x * 8 + wid;
  if (row >= a.rows) return;
  const float* x = a.X + (long)row * a.C;
  float s = 0.f, s2 = 0.f;
  for (int i = lane; i < a.C; i += 32) { float v = x[i]; s += v; s2 += v * v; }
  for (int m = 16; m; m >>= 1) { s += __shfl_xor(s, m, 32); s2 += __shfl_xor(s2, m, 32); }
  float mean = (a.flags & LF_RMS) ? 0.f : (s / a.C);
  float inv  = rsqrtf(s2 / a.C - mean * mean + 1e-5f);
  float* o = a.O + (long)row * a.C;
  const float* res = (a.flags & LF_RES) ? (a.Res + (long)row * a.C) : nullptr;
  for (int i = lane; i < a.C; i += 32) {
    float v = (x[i] - mean) * inv * a.W[i];
    if (a.flags & LF_BIAS) v += a.Bp[i];
    if (a.flags & LF_GELU) v = dev_gelu(v);
    if (a.flags & LF_RES)  v += res[i];
    o[i] = v;
  }
}

// ---------------- softmax rows (wave per row) ----------------
__global__ __launch_bounds__(256) void softmax_kernel(float* X, int rows, int len, int ld) {
  int wid = threadIdx.x >> 5, lane = threadIdx.x & 31;
  int row = blockIdx.x * 8 + wid;
  if (row >= rows) return;
  float* x = X + (long)row * ld;
  float mx = -1e30f;
  for (int i = lane; i < len; i += 32) mx = fmaxf(mx, x[i]);
  for (int m = 16; m; m >>= 1) mx = fmaxf(mx, __shfl_xor(mx, m, 32));
  float s = 0.f;
  for (int i = lane; i < len; i += 32) { float e = expf(x[i] - mx); x[i] = e; s += e; }
  for (int m = 16; m; m >>= 1) s += __shfl_xor(s, m, 32);
  float inv = 1.f / s;
  for (int i = lane; i < len; i += 32) x[i] *= inv;
}

// ---- patch-embed conv1: thread per position, all 64 channels, weights in LDS ----
__global__ __launch_bounds__(256) void conv1_kernel(const float* __restrict__ x,
                                                    const float* __restrict__ w,
                                                    _Float16* __restrict__ out, int npos) {
  __shared__ float ws[64 * 27];
  for (int i = threadIdx.x; i < 64 * 27; i += 256) ws[i] = w[i];
  __syncthreads();
  int pos = blockIdx.x * 256 + threadIdx.x;
  if (pos >= npos) return;
  const int OD=46, OH=55, OW=46, ID=91, IH=109, IW=91;
  int OSP = OD * OH * OW;
  int f  = pos / OSP;
  int rp = pos - f * OSP;
  int od = rp / (OH * OW); int r2 = rp - od * OH * OW;
  int oh = r2 / OW;        int ow = r2 - oh * OW;
  float xin[27];
  #pragma unroll
  for (int kd = 0; kd < 3; ++kd)
    #pragma unroll
    for (int kh = 0; kh < 3; ++kh)
      #pragma unroll
      for (int kw = 0; kw < 3; ++kw) {
        int id = od * 2 + kd - 1, ih = oh * 2 + kh - 1, iw = ow * 2 + kw - 1;
        float v = 0.f;
        if ((unsigned)id < (unsigned)ID && (unsigned)ih < (unsigned)IH &&
            (unsigned)iw < (unsigned)IW)
          v = x[((long)(f * ID + id) * IH + ih) * IW + iw];
        xin[kd * 9 + kh * 3 + kw] = v;
      }
  _Float16* op = out + (long)pos * 64;
  #pragma unroll
  for (int ob = 0; ob < 8; ++ob) {
    h8 pack;
    #pragma unroll
    for (int oi = 0; oi < 8; ++oi) {
      int o = ob * 8 + oi;
      float s = 0.f;
      #pragma unroll
      for (int t = 0; t < 27; ++t) s += xin[t] * ws[o * 27 + t];
      pack[oi] = (_Float16)fmaxf(s, 0.f);
    }
    ((h8*)op)[ob] = pack;
  }
}

// ---------------- conv weight repack OIDHW -> [tap][o][i] f16 ----------------
__global__ __launch_bounds__(256) void pack_convw_kernel(const float* w, _Float16* out,
                                                         int Cout, int Cin) {
  int idx = blockIdx.x * 256 + threadIdx.x;
  int total = Cout * Cin * 27;
  if (idx >= total) return;
  int t = idx / (Cout * Cin);
  int r = idx - t * Cout * Cin;
  int o = r / Cin, i = r - o * Cin;
  out[idx] = (_Float16)w[(o * Cin + i) * 27 + t];
}

// ---------------- token assembly / misc small kernels ----------------
__global__ __launch_bounds__(256) void token_kernel(const float* conv, const float* cls,
                                                    const float* pos, float* X) {
  int idx = blockIdx.x * 256 + threadIdx.x;
  if (idx >= 4 * 253 * 512) return;
  int c  = idx & 511;
  int tn = idx >> 9;
  int f = tn / 253, n = tn % 253;
  float v = (n == 0) ? cls[c] : conv[((long)f * 252 + (n - 1)) * 512 + c];
  X[idx] = v + pos[n * 512 + c];
}

__global__ __launch_bounds__(256) void extract_kernel(const float* hb, const float* tpos,
                                                      float* res) {
  int idx = blockIdx.x * 256 + threadIdx.x;
  if (idx >= 4 * 512) return;
  int f = idx >> 9, c = idx & 511;
  res[idx] = hb[(long)(f * 253) * 512 + c] + tpos[f * 512 + c];
}

// residual add staged through LDS via CDNA5 async global->LDS copy (ASYNCcnt path)
__global__ __launch_bounds__(256) void add_kernel(float* y, const float* x, int n) {
  __shared__ float stage[256];
  int i = blockIdx.x * 256 + threadIdx.x;
  if (i < n) {
    unsigned lds = (unsigned)(uintptr_t)&stage[threadIdx.x];
    unsigned long long ga = (unsigned long long)(uintptr_t)&x[i];
    asm volatile("global_load_async_to_lds_b32 %0, %1, off"
                 :: "v"(lds), "v"(ga) : "memory");
    asm volatile("s_wait_asynccnt 0" ::: "memory");
    y[i] += stage[threadIdx.x];
  }
}

// ---------------- Mamba: depthwise causal conv1d + SiLU ----------------
__global__ __launch_bounds__(256) void mamba_conv_kernel(const float* xz, const float* w,
                                                         const float* b, float* xc, int rev) {
  int idx = blockIdx.x * 256 + threadIdx.x;
  if (idx >= 4 * 1024) return;
  int ch = idx & 1023, s = idx >> 10;
  float acc = b[ch];
  #pragma unroll
  for (int j = 0; j < 4; ++j) {
    int sp = s - 3 + j;
    if (sp >= 0) {
      int l = rev ? (3 - sp) : sp;
      acc += w[ch * 4 + j] * xz[l * 2048 + ch];
    }
  }
  xc[idx] = dev_silu(acc);
}

// ---------------- Mamba selective scan (L=4, D_STATE=16) ----------------
__global__ __launch_bounds__(256) void mamba_scan_kernel(const float* xc, const float* dt,
    const float* proj, const float* A_log, const float* Dp, const float* xz,
    float* y, int rev, int accum) {
  int ch = blockIdx.x * 256 + threadIdx.x;
  if (ch >= 1024) return;
  float h[16], Ar[16];
  #pragma unroll
  for (int n = 0; n < 16; ++n) { h[n] = 0.f; Ar[n] = -expf(A_log[ch * 16 + n]); }
  for (int s = 0; s < 4; ++s) {
    float dtv = dt[s * 1024 + ch];
    float u   = xc[s * 1024 + ch];
    float yv  = 0.f;
    #pragma unroll
    for (int n = 0; n < 16; ++n) {
      float dA = expf(dtv * Ar[n]);
      h[n] = dA * h[n] + dtv * proj[s * 64 + 32 + n] * u;
      yv  += h[n] * proj[s * 64 + 48 + n];
    }
    yv += u * Dp[ch];
    int l = rev ? (3 - s) : s;
    float z = xz[l * 2048 + 1024 + ch];
    yv *= dev_silu(z);
    long oi = (long)l * 1024 + ch;
    y[oi] = accum ? (y[oi] + yv) : yv;
  }
}

// ---------------- final LN + mean + head ----------------
__global__ __launch_bounds__(128) void final_kernel(const float* res, const float* w,
    const float* b, const float* hw, const float* hb, float* out) {
  __shared__ float part[4];
  int wid = threadIdx.x >> 5, lane = threadIdx.x & 31;
  const float* x = res + wid * 512;
  float s = 0.f, s2 = 0.f;
  for (int i = lane; i < 512; i += 32) { float v = x[i]; s += v; s2 += v * v; }
  for (int m = 16; m; m >>= 1) { s += __shfl_xor(s, m, 32); s2 += __shfl_xor(s2, m, 32); }
  float mean = s / 512.f;
  float inv  = rsqrtf(s2 / 512.f - mean * mean + 1e-5f);
  float d = 0.f;
  for (int i = lane; i < 512; i += 32) {
    float v = (x[i] - mean) * inv * w[i] + b[i];
    d += v * hw[i];
  }
  for (int m = 16; m; m >>= 1) d += __shfl_xor(d, m, 32);
  if (lane == 0) part[wid] = d;
  __syncthreads();
  if (threadIdx.x == 0)
    out[0] = (part[0] + part[1] + part[2] + part[3]) * 0.25f + hb[0];
}

// ---------------- host-side helpers ----------------
static void launch_gemm(const GemmArgs& g, hipStream_t st) {
  long tiles = (long)((g.M + 15) / 16) * ((g.N + 15) / 16) * g.batch;
  int blocks = (int)((tiles + 7) / 8);
  int key = ((g.flags & GF_AHALF) ? 1 : 0) | ((g.flags & GF_BHALF) ? 2 : 0) |
            ((g.flags & GF_BKMAJ) ? 4 : 0);
  switch (key) {
    case 2:  gemm_wmma_kernel<0,1,0><<<blocks, 256, 0, st>>>(g); break;
    case 3:  gemm_wmma_kernel<1,1,0><<<blocks, 256, 0, st>>>(g); break;
    case 4:  gemm_wmma_kernel<0,0,1><<<blocks, 256, 0, st>>>(g); break;
    default: gemm_wmma_kernel<0,0,0><<<blocks, 256, 0, st>>>(g); break;
  }
}

static void gemmL(hipStream_t st, const void* A, const void* B, float* C, const float* bias,
                  int M, int N, int K, int lda, int ldb, int ldc, int flags, float scale = 1.f,
                  int batch = 1, int binner = 1,
                  long sAo = 0, long sAi = 0, long sBo = 0, long sBi = 0,
                  long sCo = 0, long sCi = 0) {
  GemmArgs g{};
  g.A = A; g.B = B; g.C = C; g.bias = bias;
  g.M = M; g.N = N; g.Kb = K;
  g.lda = lda; g.ldb = ldb; g.ldc = ldc;
  g.batch = batch; g.binner = binner;
  g.sAo = sAo; g.sAi = sAi; g.sBo = sBo; g.sBi = sBi; g.sCo = sCo; g.sCi = sCi;
  g.scale = scale; g.flags = flags; g.conv = 0;
  launch_gemm(g, st);
}

static void convL(hipStream_t st, const void* X, const _Float16* Wp, float* C,
                  const float* bias, int frames, int ID, int IH, int IW,
                  int OD, int OH, int OW, int stride, int Cin, int Cout, int extraFlags) {
  GemmArgs g{};
  g.A = X; g.B = Wp; g.C = C; g.bias = bias;
  g.M = frames * OD * OH * OW; g.N = Cout; g.Kb = Cin;
  g.lda = Cin; g.ldb = Cin; g.ldc = Cout;
  g.batch = 1; g.binner = 1; g.scale = 1.f;
  g.flags = GF_BHALF | extraFlags;
  g.conv = 1; g.frames = frames;
  g.ID = ID; g.IH = IH; g.IW = IW; g.OD = OD; g.OH = OH; g.OW = OW; g.cstride = stride;
  launch_gemm(g, st);
}

static void lnL(hipStream_t st, const float* X, const float* W, const float* Bp,
                const float* Res, float* O, int rows, int C, int flags) {
  LnArgs a{X, W, Bp, Res, O, rows, C, flags};
  ln_kernel<<<(rows + 7) / 8, 256, 0, st>>>(a);
}

// packed conv weight offsets (in _Float16 elements)
static const long HP_PE2   = 0;
static const long HP_CB0W1 = 221184;
static const long HP_CB0W2 = 663552;
static const long HP_DS0   = 1105920;
static const long HP_CB1W1 = 1990656;
static const long HP_CB1W2 = 3760128;
static const long HP_DS1   = 5529600;

// workspace byte offsets (peak usage ~143 MB; requires ws_size >= 160 MB)
#define OFF_C1  (20ll << 20)   // conv1 out f16 (59.6MB), reused as T1/D0 later
#define OFF_A2  (80ll << 20)   // conv2 out fp32 (30.3MB)
#define OFF_T2  (112ll << 20)  // cb0 second conv out (30.3MB)
#define OFF_B1  (32ll << 20)
#define OFF_B2  (48ll << 20)
#define OFF_T0  (64ll << 20)
#define OFF_X   (68ll << 20)
#define OFF_HB  (72ll << 20)
#define OFF_QKV (76ll << 20)
#define OFF_ATT (84ll << 20)
#define OFF_OB  (88ll << 20)
#define OFF_MLP (92ll << 20)
#define OFF_RES (104ll << 20)
#define OFF_HID (OFF_RES + (64ll  << 10))
#define OFF_RN  (OFF_RES + (128ll << 10))
#define OFF_XZ  (OFF_RES + (192ll << 10))
#define OFF_XC  (OFF_RES + (256ll << 10))
#define OFF_PRJ (OFF_RES + (320ll << 10))
#define OFF_DT  (OFF_RES + (384ll << 10))
#define OFF_YS  (OFF_RES + (448ll << 10))

extern "C" void kernel_launch(void* const* d_in, const int* in_sizes, int n_in,
                              void* d_out, int out_size, void* d_ws, size_t ws_size,
                              hipStream_t stream) {
  (void)in_sizes; (void)n_in; (void)out_size; (void)ws_size;
  auto F = [&](int i) { return (const float*)d_in[i]; };
  char* w8 = (char*)d_ws;
  _Float16* WP = (_Float16*)w8;

  // ---- pack conv weights (tap-major, f16) ----
  auto packW = [&](int inIdx, long hoff, int Cout, int Cin) {
    int total = Cout * Cin * 27;
    pack_convw_kernel<<<(total + 255) / 256, 256, 0, stream>>>(F(inIdx), WP + hoff, Cout, Cin);
  };
  packW(2,  HP_PE2,   128, 64);
  packW(3,  HP_CB0W1, 128, 128);
  packW(7,  HP_CB0W2, 128, 128);
  packW(11, HP_DS0,   256, 128);
  packW(12, HP_CB1W1, 256, 256);
  packW(16, HP_CB1W2, 256, 256);
  packW(20, HP_DS1,   512, 256);

  // ---- patch embed ----
  _Float16* C1 = (_Float16*)(w8 + OFF_C1);
  const int NPOS1 = 4 * 46 * 55 * 46;  // 465520
  conv1_kernel<<<(NPOS1 + 255) / 256, 256, 0, stream>>>(F(0), F(1), C1, NPOS1);
  float* A2 = (float*)(w8 + OFF_A2);
  convL(stream, C1, WP + HP_PE2, A2, nullptr, 4, 46,55,46, 23,28,23, 2, 64, 128,
        GF_AHALF | GF_RELU);

  // ---- conv block 0 (128ch @ 23x28x23) ----
  float* T1 = (float*)(w8 + OFF_C1);   // reuse conv1 region
  float* T2 = (float*)(w8 + OFF_T2);
  convL(stream, A2, WP + HP_CB0W1, T1, F(4), 4, 23,28,23, 23,28,23, 1, 128, 128, GF_BIAS);
  lnL(stream, T1, F(5), F(6), nullptr, T1, 59248, 128, LF_BIAS | LF_GELU);
  convL(stream, T1, WP + HP_CB0W2, T2, F(8), 4, 23,28,23, 23,28,23, 1, 128, 128, GF_BIAS);
  lnL(stream, T2, F(9), F(10), A2, A2, 59248, 128, LF_BIAS | LF_RES);

  // ---- downsample 0 ----
  float* D0 = (float*)(w8 + OFF_C1);   // reuse again (T1 dead)
  convL(stream, A2, WP + HP_DS0, D0, nullptr, 4, 23,28,23, 12,14,12, 2, 128, 256, 0);

  // ---- conv block 1 (256ch @ 12x14x12) ----
  float* B1 = (float*)(w8 + OFF_B1);
  float* B2 = (float*)(w8 + OFF_B2);
  convL(stream, D0, WP + HP_CB1W1, B1, F(13), 4, 12,14,12, 12,14,12, 1, 256, 256, GF_BIAS);
  lnL(stream, B1, F(14), F(15), nullptr, B1, 8064, 256, LF_BIAS | LF_GELU);
  convL(stream, B1, WP + HP_CB1W2, B2, F(17), 4, 12,14,12, 12,14,12, 1, 256, 256, GF_BIAS);
  lnL(stream, B2, F(18), F(19), D0, D0, 8064, 256, LF_BIAS | LF_RES);

  // ---- downsample 1 -> tokens ----
  float* T0 = (float*)(w8 + OFF_T0);
  convL(stream, D0, WP + HP_DS1, T0, nullptr, 4, 12,14,12, 6,7,6, 2, 256, 512, 0);
  float* X = (float*)(w8 + OFF_X);
  token_kernel<<<(4 * 253 * 512 + 255) / 256, 256, 0, stream>>>(T0, F(21), F(22), X);

  // ---- 12 attention blocks ----
  float* HBp = (float*)(w8 + OFF_HB);
  float* QKV = (float*)(w8 + OFF_QKV);
  float* ATT = (float*)(w8 + OFF_ATT);
  float* OB  = (float*)(w8 + OFF_OB);
  float* MLP = (float*)(w8 + OFF_MLP);
  for (int b = 0; b < 12; ++b) {
    int ba = 24 + b * 12;
    lnL(stream, X, F(ba + 0), F(ba + 1), nullptr, HBp, 1012, 512, LF_BIAS);
    gemmL(stream, HBp, F(ba + 2), QKV, F(ba + 3), 1012, 1536, 512, 512, 512, 1536, GF_BIAS);
    // S = q k^T * hd^-0.5   (batch over frame x head)
    gemmL(stream, QKV, (const float*)QKV + 512, ATT, nullptr, 253, 253, 256,
          1536, 1536, 256, 0, 0.0625f, 8, 2,
          253ll * 1536, 256, 253ll * 1536, 256, 2ll * 253 * 256, 253ll * 256);
    softmax_kernel<<<(2024 + 7) / 8, 256, 0, stream>>>(ATT, 2024, 253, 256);
    // O = att @ V
    gemmL(stream, ATT, (const float*)QKV + 1024, OB, nullptr, 253, 256, 253,
          256, 1536, 512, GF_BKMAJ, 1.f, 8, 2,
          2ll * 253 * 256, 253ll * 256, 253ll * 1536, 256, 253ll * 512, 256);
    gemmL(stream, OB, F(ba + 4), X, F(ba + 5), 1012, 512, 512, 512, 512, 512,
          GF_BIAS | GF_ADDTO);
    lnL(stream, X, F(ba + 6), F(ba + 7), nullptr, HBp, 1012, 512, LF_BIAS);
    gemmL(stream, HBp, F(ba + 8), MLP, F(ba + 9), 1012, 2048, 512, 512, 512, 2048,
          GF_BIAS | GF_GELU);
    gemmL(stream, MLP, F(ba + 10), X, F(ba + 11), 1012, 512, 2048, 2048, 2048, 512,
          GF_BIAS | GF_ADDTO);
  }

  // ---- final attn norm + temporal token extraction ----
  lnL(stream, X, F(168), F(169), nullptr, HBp, 1012, 512, LF_BIAS);
  float* RES = (float*)(w8 + OFF_RES);
  extract_kernel<<<(2048 + 255) / 256, 256, 0, stream>>>(HBp, F(23), RES);

  // ---- 8 Mamba blocks ----
  float* HID = (float*)(w8 + OFF_HID);
  float* RN  = (float*)(w8 + OFF_RN);
  float* XZ  = (float*)(w8 + OFF_XZ);
  float* XC  = (float*)(w8 + OFF_XC);
  float* PRJ = (float*)(w8 + OFF_PRJ);
  float* DT  = (float*)(w8 + OFF_DT);
  float* YS  = (float*)(w8 + OFF_YS);
  for (int mb = 0; mb < 8; ++mb) {
    int ba = 170 + mb * 17;
    if (mb > 0) add_kernel<<<8, 256, 0, stream>>>(RES, HID, 2048);
    lnL(stream, RES, F(ba + 0), nullptr, nullptr, RN, 4, 512, LF_RMS);
    gemmL(stream, RN, F(ba + 1), XZ, nullptr, 4, 2048, 512, 512, 512, 2048, 0);
    for (int dir = 0; dir < 2; ++dir) {
      int mo = ba + 2 + dir * 7;
      mamba_conv_kernel<<<16, 256, 0, stream>>>(XZ, F(mo + 0), F(mo + 1), XC, dir);
      gemmL(stream, XC, F(mo + 2), PRJ, nullptr, 4, 64, 1024, 1024, 1024, 64, 0);
      gemmL(stream, PRJ, F(mo + 3), DT, F(mo + 4), 4, 1024, 32, 64, 32, 1024,
            GF_BIAS | GF_SOFTPLUS);
      mamba_scan_kernel<<<4, 256, 0, stream>>>(XC, DT, PRJ, F(mo + 5), F(mo + 6),
                                               XZ, YS, dir, dir);
    }
    gemmL(stream, YS, F(ba + 16), HID, nullptr, 4, 512, 1024, 1024, 1024, 512, 0);
  }
  add_kernel<<<8, 256, 0, stream>>>(RES, HID, 2048);

  // ---- final norm + mean + head ----
  final_kernel<<<1, 128, 0, stream>>>(RES, F(306), F(307), F(308), F(309), (float*)d_out);
}
